// Transformer_2869038154522
// MI455X (gfx1250) — compile-verified, matrix operand order checked by reading, not run on hardware
//
#include <hip/hip_runtime.h>
#include <hip/hip_bf16.h>
#include <math.h>

// ---------------------------------------------------------------------------
// Transformer encoder forward for MI455X (gfx1250, wave32, WMMA + TDM).
// GEMMs: v_wmma_f32_16x16x32_f16 with Tensor-Data-Mover double-buffered LDS
// staging (tensor_load_to_lds + s_wait_tensorcnt).
// ---------------------------------------------------------------------------

typedef __attribute__((ext_vector_type(16))) _Float16 v16h;
typedef __attribute__((ext_vector_type(8)))  float    v8f;

#define TB 16
#define TT 256
#define TC 1024
#define TH 16
#define TL 8
#define DH 64           // head dim
#define LN_EPS 1e-5f

// ---- WMMA fragment index math (ISA 7.12.2, 16-bit A 16x32 layout) ----------
// A fragment (16x32, MxK): lane&15 = M row; lanes>=16 shift K by +8 within a
// 16-wide K group; v16h element s -> VGPR s/2, halfword s%2.
// B fragment (32x16, KxN) uses the symmetric layout with lane&15 = N col.
__device__ __forceinline__ int frag_k_local(int lane, int s) {
    int v    = s >> 1;          // vgpr index 0..7
    int hw   = s & 1;           // halfword
    int half = (lane >> 4) & 1; // lane group
    return ((v & 4) ? 16 : 0) + half * 8 + (v & 3) * 2 + hw;
}

// Pack f32 source into f16 WMMA fragments.
// Fragment storage: frag[rowTile][kTile][lane(32)][slot(16)] contiguous f16
// (one 16x32 / 32x16 tile = 512 f16 = 1KB). grid.y/z apply batch offsets.
__global__ __launch_bounds__(256) void pack_frag_kernel(
    const float* __restrict__ src, _Float16* __restrict__ dst,
    int rows, int K, long sRow, long sK, long sY, long sZ, long dY, long dZ)
{
    long idx = (long)blockIdx.x * 256 + threadIdx.x;
    long total = (long)rows * K;
    if (idx >= total) return;
    src += (long)blockIdx.y * sY + (long)blockIdx.z * sZ;
    dst += (long)blockIdx.y * dY + (long)blockIdx.z * dZ;

    int  s    = (int)(idx & 15);
    int  lane = (int)((idx >> 4) & 31);
    long tile = idx >> 9;
    int  ktn  = K >> 5;
    int  kt   = (int)(tile % ktn);
    long rt   = tile / ktn;
    long r    = rt * 16 + (lane & 15);
    long k    = (long)kt * 32 + frag_k_local(lane, s);
    dst[idx] = (_Float16)src[r * sRow + k * sK];
}

// ---- Tensor Data Mover: 2D strided fragment-slice load into LDS ------------
// Loads `tiles` tiles of 512 f16 (1KB) each, source stride `strideElems` f16,
// into contiguous LDS at lptr. D# per ISA ch.8: group0 {count, lds_addr,
// global_addr, type=2}; group1 {data_size=2B, tensor_dim0=512, tensor_dim1=
// tiles, tile_dim0=512, tile_dim1=tiles, tensor_dim0_stride=strideElems}.
// This toolchain exposes the 6-arg builtin (g0, g1, g2, g3, g4, cpol).
__device__ __forceinline__ void tdm_load_frag_slice(
    _Float16* lptr, const _Float16* gptr, unsigned tiles, unsigned strideElems,
    int lane)
{
#if __has_builtin(__builtin_amdgcn_tensor_load_to_lds)
    typedef unsigned int v4u __attribute__((ext_vector_type(4)));
    typedef int v8i __attribute__((ext_vector_type(8)));
    typedef int v4i __attribute__((ext_vector_type(4)));
    (void)lane;
    unsigned long long ga = (unsigned long long)gptr;
    unsigned lds_byte = (unsigned)(unsigned long long)lptr;  // low 32 = LDS offset
    v4u g0;
    g0[0] = 1u;                                        // count = 1 valid D#
    g0[1] = lds_byte;                                  // lds_addr
    g0[2] = (unsigned)(ga & 0xffffffffu);              // global_addr[31:0]
    g0[3] = (unsigned)((ga >> 32) & 0x1ffffffu) | (2u << 30);  // [56:32]|type=2
    v8i g1;
    g1[0] = (int)(1u << 16);                           // data_size = 2 bytes
    g1[1] = (int)(512u << 16);                         // tensor_dim0 = 512
    g1[2] = (int)((tiles & 0xffffu) << 16);            // tensor_dim1 = tiles
    g1[3] = (int)(512u << 16);                         // tile_dim0 = 512
    g1[4] = (int)(tiles & 0xffffu);                    // tile_dim1 = tiles
    g1[5] = (int)strideElems;                          // tensor_dim0_stride lo
    g1[6] = 0;                                         // stride hi / dim1_stride
    g1[7] = 0;
    v4i z4 = {0, 0, 0, 0};                             // groups 2/3 unused (2D)
    v8i z8 = {0, 0, 0, 0, 0, 0, 0, 0};
    __builtin_amdgcn_tensor_load_to_lds(g0, g1, z4, z4, z8, 0);
#else
    // Fallback: wave-cooperative synchronous copy (32 lanes, 2 x b128 / tile).
    for (unsigned t = 0; t < tiles; ++t) {
        const uint4* s = (const uint4*)(gptr + (long)t * strideElems);
        uint4*       d = (uint4*)(lptr + t * 512);
        d[lane]      = s[lane];
        d[lane + 32] = s[lane + 32];
    }
#endif
}

__device__ __forceinline__ void tdm_wait_stage(int more_inflight) {
#if __has_builtin(__builtin_amdgcn_s_wait_tensorcnt)
    if (more_inflight) __builtin_amdgcn_s_wait_tensorcnt(2);
    else               __builtin_amdgcn_s_wait_tensorcnt(0);
#else
    (void)more_inflight;
#endif
}

// Batched WMMA GEMM:  out[M,N] = epilogue(A[M,K] @ B[K,N] + bias) (+ res)
// Block = 8 waves = 128(M) x 64(N) tile. TDM stages 8 A-tiles + 4 B-tiles per
// K-step into double-buffered LDS; each wave runs 4 wmma per K-step.
// Requires M % 128 == 0, N % 64 == 0, K % 32 == 0.
__global__ __launch_bounds__(256) void gemm_wmma_kernel(
    const _Float16* __restrict__ A, const _Float16* __restrict__ Bf,
    const float* __restrict__ bias, const float* __restrict__ res,
    float* __restrict__ out, int M, int N, int K, int act,
    long aBatch, long bBatch, long oBatch, long rBatch)
{
    __shared__ __align__(32) _Float16 sA[2][8 * 512];   // 16 KB
    __shared__ __align__(32) _Float16 sB[2][4 * 512];   //  8 KB

    A   += (long)blockIdx.z * aBatch;
    Bf  += (long)blockIdx.z * bBatch;
    out += (long)blockIdx.z * oBatch;
    const float* resp = res ? (res + (long)blockIdx.z * rBatch) : nullptr;

    int lane = threadIdx.x & 31;
    int wave = threadIdx.x >> 5;
    int mT0  = blockIdx.y * 8;
    int nT0  = blockIdx.x << 2;
    int ktn  = K >> 5;
    unsigned strideE = (unsigned)(ktn * 512);

    const _Float16* Aslice = A  + (long)mT0 * ktn * 512;   // +kt*512 per stage
    const _Float16* Bslice = Bf + (long)nT0 * ktn * 512;

    if (wave == 0) {
        tdm_load_frag_slice(sA[0], Aslice, 8, strideE, lane);
        tdm_load_frag_slice(sB[0], Bslice, 4, strideE, lane);
        if (ktn > 1) {
            tdm_load_frag_slice(sA[1], Aslice + 512, 8, strideE, lane);
            tdm_load_frag_slice(sB[1], Bslice + 512, 4, strideE, lane);
        }
    }

    v8f acc0 = {}, acc1 = {}, acc2 = {}, acc3 = {};
    for (int kt = 0; kt < ktn; ++kt) {
        int buf = kt & 1;
        if (wave == 0) tdm_wait_stage(kt + 1 < ktn);
        __syncthreads();

        const int lo = lane << 4;
        v16h a  = *(const v16h*)(sA[buf] + (wave << 9) + lo);
        v16h b0 = *(const v16h*)(sB[buf] +           lo);
        v16h b1 = *(const v16h*)(sB[buf] +  512    + lo);
        v16h b2 = *(const v16h*)(sB[buf] + 1024    + lo);
        v16h b3 = *(const v16h*)(sB[buf] + 1536    + lo);
        acc0 = __builtin_amdgcn_wmma_f32_16x16x32_f16(false, a, false, b0, (short)0, acc0, false, false);
        acc1 = __builtin_amdgcn_wmma_f32_16x16x32_f16(false, a, false, b1, (short)0, acc1, false, false);
        acc2 = __builtin_amdgcn_wmma_f32_16x16x32_f16(false, a, false, b2, (short)0, acc2, false, false);
        acc3 = __builtin_amdgcn_wmma_f32_16x16x32_f16(false, a, false, b3, (short)0, acc3, false, false);

        __syncthreads();                       // all waves done reading buf
        if (wave == 0 && kt + 2 < ktn) {       // refill freed buffer
            tdm_load_frag_slice(sA[buf], Aslice + (long)(kt + 2) * 512, 8, strideE, lane);
            tdm_load_frag_slice(sB[buf], Bslice + (long)(kt + 2) * 512, 4, strideE, lane);
        }
    }

    // C/D layout: VGPR r -> rows r (lanes 0-15) and r+8 (lanes 16-31), col = lane&15
    int mT    = mT0 + wave;
    int half8 = (lane >> 4) << 3;
    int ncb   = lane & 15;
    v8f accs[4] = {acc0, acc1, acc2, acc3};
#pragma unroll
    for (int j = 0; j < 4; ++j) {
        int   ncol = (nT0 + j) * 16 + ncb;
        float bl   = bias ? bias[ncol] : 0.f;
#pragma unroll
        for (int r = 0; r < 8; ++r) {
            long  row = (long)mT * 16 + half8 + r;
            float val = accs[j][r] + bl;
            if (act) val = 0.5f * val * (1.f + erff(val * 0.70710678118654752f));
            if (resp) val += resp[row * N + ncol];
            out[row * N + ncol] = val;
        }
    }
}

// x = emb[tok] + pos
__global__ __launch_bounds__(256) void embed_kernel(
    const int* __restrict__ tokens, const float* __restrict__ emb,
    const float* __restrict__ pos, float* __restrict__ x)
{
    long idx = (long)blockIdx.x * 256 + threadIdx.x;
    if (idx >= (long)TB * TT * TC) return;
    int  c  = (int)(idx & (TC - 1));
    long bt = idx >> 10;
    int  t  = (int)(bt & (TT - 1));
    long tok = tokens[bt];
    x[idx] = emb[tok * TC + c] + pos[(long)t * TC + c];
}

// Row LayerNorm over C=1024, one block per row.
__global__ __launch_bounds__(256) void ln_kernel(
    const float* __restrict__ x, const float* __restrict__ g,
    const float* __restrict__ b, float* __restrict__ y)
{
    __shared__ float ssum[256];
    __shared__ float ssq[256];
    long row = blockIdx.x;
    const float* xr = x + row * TC;
    float s = 0.f, q = 0.f;
    for (int i = threadIdx.x; i < TC; i += 256) { float v = xr[i]; s += v; q += v * v; }
    ssum[threadIdx.x] = s; ssq[threadIdx.x] = q;
    __syncthreads();
    for (int off = 128; off > 0; off >>= 1) {
        if (threadIdx.x < (unsigned)off) {
            ssum[threadIdx.x] += ssum[threadIdx.x + off];
            ssq[threadIdx.x]  += ssq[threadIdx.x + off];
        }
        __syncthreads();
    }
    float mean = ssum[0] * (1.f / TC);
    float var  = ssq[0] * (1.f / TC) - mean * mean;
    float inv  = rsqrtf(var + LN_EPS);
    for (int i = threadIdx.x; i < TC; i += 256)
        y[row * TC + i] = (xr[i] - mean) * inv * g[i] + b[i];
}

// In-place softmax over rows of S[B,H,T,T] with scale + pad-mask bias.
__global__ __launch_bounds__(256) void softmax_kernel(
    float* __restrict__ S, const int* __restrict__ tokens, float scale)
{
    __shared__ float red[256];
    long row = blockIdx.x;                 // 0 .. B*H*T-1
    int  b   = (int)(row >> 12);           // / (H*T)
    long base = row * TT;
    int  j   = threadIdx.x;
    float v = S[base + j] * scale + ((tokens[(long)b * TT + j] == 0) ? -10000.f : 0.f);
    red[j] = v;
    __syncthreads();
    for (int off = 128; off > 0; off >>= 1) {
        if (j < off) red[j] = fmaxf(red[j], red[j + off]);
        __syncthreads();
    }
    float mx = red[0];
    __syncthreads();
    float e = __expf(v - mx);
    red[j] = e;
    __syncthreads();
    for (int off = 128; off > 0; off >>= 1) {
        if (j < off) red[j] += red[j + off];
        __syncthreads();
    }
    S[base + j] = e / red[0];
}

// y[b,h,t,d] -> ycat[b,t,h*64+d]
__global__ __launch_bounds__(256) void permute_y_kernel(
    const float* __restrict__ yb, float* __restrict__ yc)
{
    long idx = (long)blockIdx.x * 256 + threadIdx.x;
    if (idx >= (long)TB * TT * TC) return;
    int  c  = (int)(idx & (TC - 1));
    long bt = idx >> 10;
    int  t  = (int)(bt & (TT - 1));
    long b  = bt >> 8;
    int  h  = c >> 6, d = c & 63;
    yc[idx] = yb[(((b * TH + h) * TT) + t) * DH + d];
}

// ---------------------------------------------------------------------------
extern "C" void kernel_launch(void* const* d_in, const int* in_sizes, int n_in,
                              void* d_out, int out_size, void* d_ws, size_t ws_size,
                              hipStream_t stream) {
    (void)in_sizes; (void)n_in; (void)out_size; (void)ws_size;
    const long M  = (long)TB * TT;   // 4096 tokens
    const long MC = M * TC;          // 4M elements

    const int*   tokens = (const int*)d_in[0];
    const float* emb    = (const float*)d_in[2];
    const float* pos    = (const float*)d_in[3];
    const float* ln1g   = (const float*)d_in[4];
    const float* ln1b   = (const float*)d_in[5];
    const float* Wq     = (const float*)d_in[6];
    const float* bq     = (const float*)d_in[7];
    const float* Wk     = (const float*)d_in[8];
    const float* bk     = (const float*)d_in[9];
    const float* Wv     = (const float*)d_in[10];
    const float* bv     = (const float*)d_in[11];
    const float* Wo     = (const float*)d_in[12];
    const float* bo     = (const float*)d_in[13];
    const float* ln2g   = (const float*)d_in[14];
    const float* ln2b   = (const float*)d_in[15];
    const float* W1     = (const float*)d_in[16];
    const float* b1     = (const float*)d_in[17];
    const float* W2     = (const float*)d_in[18];
    const float* b2     = (const float*)d_in[19];
    const float* lnfg   = (const float*)d_in[20];
    const float* lnfb   = (const float*)d_in[21];

    // ---- workspace carve-up (float units) ----
    float* ws = (float*)d_ws;
    long off = 0;
    auto allocF = [&](long n) { float* p = ws + off; off += n; return p; };
    float* x     = allocF(MC);
    float* h     = allocF(MC);     // ln1 out / attn residual / ln2 out
    float* q     = allocF(MC);
    float* k     = allocF(MC);
    float* v     = allocF(MC);
    float* ybhtd = allocF(MC);
    float* ycat  = allocF(MC);
    float* m1    = allocF(MC);
    float* S     = allocF((long)TB * TH * TT * TT);     // scores / probs
    _Float16* afrag = (_Float16*)(ws + off); off += MC / 2;                 // activations, f16
    _Float16* wfrag = (_Float16*)(ws + off); off += (long)TC * TC / 2;      // one weight, f16
    _Float16* qfrag = (_Float16*)(ws + off); off += MC / 2;
    _Float16* kfrag = (_Float16*)(ws + off); off += MC / 2;
    _Float16* vfrag = (_Float16*)(ws + off); off += MC / 2;
    _Float16* pfrag = (_Float16*)(ws + off); off += (long)TB * TH * TT * TT / 2;

    auto pack = [&](const float* src, _Float16* dst, int rows, int K,
                    long sRow, long sK, long sY, long sZ, long dY, long dZ,
                    int gy, int gz) {
        long total = (long)rows * K;
        dim3 g((unsigned)((total + 255) / 256), (unsigned)gy, (unsigned)gz);
        pack_frag_kernel<<<g, 256, 0, stream>>>(src, dst, rows, K, sRow, sK, sY, sZ, dY, dZ);
    };
    auto gemm = [&](const _Float16* A, const _Float16* Bf, const float* bias,
                    const float* res, float* out, int Mm, int Nn, int Kk, int act,
                    long aB, long bB, long oB, long rB, int batches) {
        dim3 g((unsigned)(Nn / 64), (unsigned)(Mm / 128), (unsigned)batches);
        gemm_wmma_kernel<<<g, 256, 0, stream>>>(A, Bf, bias, res, out,
                                                Mm, Nn, Kk, act, aB, bB, oB, rB);
    };

    const unsigned elemBlocks = (unsigned)((MC + 255) / 256);
    embed_kernel<<<elemBlocks, 256, 0, stream>>>(tokens, emb, pos, x);

    for (int l = 0; l < TL; ++l) {
        const float* Wq_l = Wq + (long)l * TC * TC;
        const float* Wk_l = Wk + (long)l * TC * TC;
        const float* Wv_l = Wv + (long)l * TC * TC;
        const float* Wo_l = Wo + (long)l * TC * TC;
        const float* W1_l = W1 + (long)l * TC * TC;
        const float* W2_l = W2 + (long)l * TC * TC;

        // h = LN1(x)
        ln_kernel<<<(unsigned)M, 256, 0, stream>>>(x, ln1g + l * TC, ln1b + l * TC, h);

        // Q/K/V projections (WMMA)
        pack(h, afrag, (int)M, TC, TC, 1, 0, 0, 0, 0, 1, 1);
        pack(Wq_l, wfrag, TC, TC, 1, TC, 0, 0, 0, 0, 1, 1);
        gemm(afrag, wfrag, bq + l * TC, nullptr, q, (int)M, TC, TC, 0, 0, 0, 0, 0, 1);
        pack(Wk_l, wfrag, TC, TC, 1, TC, 0, 0, 0, 0, 1, 1);
        gemm(afrag, wfrag, bk + l * TC, nullptr, k, (int)M, TC, TC, 0, 0, 0, 0, 0, 1);
        pack(Wv_l, wfrag, TC, TC, 1, TC, 0, 0, 0, 0, 1, 1);
        gemm(afrag, wfrag, bv + l * TC, nullptr, v, (int)M, TC, TC, 0, 0, 0, 0, 0, 1);

        // Attention per (b,h): S = q@k^T (WMMA), softmax, Y = P@v (WMMA)
        pack(q, qfrag, TT, DH, TC, 1, (long)TT * TC, DH, (long)TH * TT * DH, (long)TT * DH, TB, TH);
        pack(k, kfrag, TT, DH, TC, 1, (long)TT * TC, DH, (long)TH * TT * DH, (long)TT * DH, TB, TH);
        pack(v, vfrag, DH, TT, 1, TC, (long)TT * TC, DH, (long)TH * DH * TT, (long)DH * TT, TB, TH);
        gemm(qfrag, kfrag, nullptr, nullptr, S, TT, TT, DH, 0,
             (long)TT * DH, (long)TT * DH, (long)TT * TT, 0, TB * TH);
        softmax_kernel<<<(unsigned)(TB * TH * TT), 256, 0, stream>>>(S, tokens, 0.125f);
        pack(S, pfrag, TT, TT, TT, 1, (long)TH * TT * TT, (long)TT * TT,
             (long)TH * TT * TT, (long)TT * TT, TB, TH);
        gemm(pfrag, vfrag, nullptr, nullptr, ybhtd, TT, DH, TT, 0,
             (long)TT * TT, (long)DH * TT, (long)TT * DH, 0, TB * TH);
        permute_y_kernel<<<elemBlocks, 256, 0, stream>>>(ybhtd, ycat);

        // x = h + (y @ Wo + bo)
        pack(ycat, afrag, (int)M, TC, TC, 1, 0, 0, 0, 0, 1, 1);
        pack(Wo_l, wfrag, TC, TC, 1, TC, 0, 0, 0, 0, 1, 1);
        gemm(afrag, wfrag, bo + l * TC, h, x, (int)M, TC, TC, 0, 0, 0, 0, 0, 1);

        // MLP: m = gelu(h2@W1+b1); x = x + gelu(m@W2+b2)
        ln_kernel<<<(unsigned)M, 256, 0, stream>>>(x, ln2g + l * TC, ln2b + l * TC, h);
        pack(h, afrag, (int)M, TC, TC, 1, 0, 0, 0, 0, 1, 1);
        pack(W1_l, wfrag, TC, TC, 1, TC, 0, 0, 0, 0, 1, 1);
        gemm(afrag, wfrag, b1 + l * TC, nullptr, m1, (int)M, TC, TC, 1, 0, 0, 0, 0, 1);
        pack(m1, afrag, (int)M, TC, TC, 1, 0, 0, 0, 0, 1, 1);
        pack(W2_l, wfrag, TC, TC, 1, TC, 0, 0, 0, 0, 1, 1);
        gemm(afrag, wfrag, b2 + l * TC, x, x, (int)M, TC, TC, 1, 0, 0, 0, 0, 1);
    }

    ln_kernel<<<(unsigned)M, 256, 0, stream>>>(x, lnfg, lnfb, (float*)d_out);
}
